// AttentionBlock_38895223832691
// MI455X (gfx1250) — matrix operand order repaired and checked by reading, not compiled, not run on hardware
//
#include <hip/hip_runtime.h>
#include <hip/hip_bf16.h>

typedef __bf16 bf16;
typedef __attribute__((ext_vector_type(16))) __bf16 v16bf;
typedef __attribute__((ext_vector_type(8)))  __bf16 v8bf;
typedef __attribute__((ext_vector_type(8)))  float  v8f;
typedef __attribute__((ext_vector_type(4)))  unsigned int u32x4;
typedef __attribute__((ext_vector_type(8)))  int i32x8;
typedef __attribute__((ext_vector_type(4)))  int i32x4;

#if defined(__has_builtin)
#  if __has_builtin(__builtin_amdgcn_tensor_load_to_lds)
#    define HAVE_TDM 1
#  else
#    define HAVE_TDM 0
#  endif
#else
#  define HAVE_TDM 0
#endif

__device__ __forceinline__ void wait_tensorcnt_0() {
#if __has_builtin(__builtin_amdgcn_s_wait_tensorcnt)
    __builtin_amdgcn_s_wait_tensorcnt(0);
#else
    asm volatile("s_wait_tensorcnt 0x0" ::: "memory");
#endif
}
__device__ __forceinline__ void wait_tensorcnt_2() {
#if __has_builtin(__builtin_amdgcn_s_wait_tensorcnt)
    __builtin_amdgcn_s_wait_tensorcnt(2);
#else
    asm volatile("s_wait_tensorcnt 0x2" ::: "memory");
#endif
}

// ---------------------------------------------------------------------------
// TDM: DMA one 2D bf16 tile (tile_d1 rows x tile_d0 elems) from a row-major
// global tensor (row stride ld_elems) into LDS at lds_dst, inserting
// pad_amt(+1) DWORDs of LDS padding every 2^(pad_int+1) DWORDs so the LDS
// layout matches the bank-conflict-padded row stride used by the WMMA frags.
// ---------------------------------------------------------------------------
__device__ __forceinline__ void tdm_load_tile(
    const bf16* __restrict__ g, int ld_elems, const bf16* lds_dst,
    unsigned tile_d0, unsigned tile_d1, unsigned pad_int, unsigned pad_amt)
{
#if HAVE_TDM
    unsigned long long ga = (unsigned long long)(size_t)g;
    unsigned lds_off = (unsigned)(size_t)lds_dst;   // LDS aperture: low 32 bits

    u32x4 g0;
    g0[0] = 1u;                                         // count=1, user D#
    g0[1] = lds_off;                                    // lds_addr
    g0[2] = (unsigned)ga;                               // global_addr[31:0]
    g0[3] = (unsigned)((ga >> 32) & 0x01FFFFFFu)        // global_addr[56:32]
          | (2u << 30);                                 // type = image

    i32x8 g1;
    g1[0] = (int)((1u << 16)                            // data_size = 2 bytes
                | (1u << 20)                            // pad_enable
                | (pad_int << 22) | (pad_amt << 25));   // pad interval/amount
    unsigned d0 = (unsigned)ld_elems;                   // tensor_dim0 = ld
    g1[1] = (int)((d0 & 0xFFFFu) << 16);                // tensor_dim0[15:0]
    g1[2] = (int)((d0 >> 16) & 0xFFFFu);                // dim0 hi | dim1 lo(=0)
    g1[3] = (int)(0x10u | (tile_d0 << 16));             // dim1 hi (1M) | tile_dim0
    g1[4] = (int)(tile_d1 & 0xFFFFu);                   // tile_dim1 | tile_dim2=0
    g1[5] = (int)d0;                                    // tensor_dim0_stride lo32
    g1[6] = 0;                                          // stride hi | dim1_stride lo
    g1[7] = 0;

    i32x4 zz = {0, 0, 0, 0};
#  if __clang_major__ >= 23
    i32x8 z8 = {0, 0, 0, 0, 0, 0, 0, 0};
    __builtin_amdgcn_tensor_load_to_lds(g0, g1, zz, zz, z8, 0);
#  else
    __builtin_amdgcn_tensor_load_to_lds(g0, g1, zz, zz, 0);
#  endif
#else
    (void)g; (void)ld_elems; (void)lds_dst;
    (void)tile_d0; (void)tile_d1; (void)pad_int; (void)pad_amt;
#endif
}

// ---------------------------------------------------------------------------
// Tiled bf16 WMMA GEMM:  C[M x N] = A[M x K] * B[K x N] (+ bias[m])
//   BT      : B operand supplied as (N x K) row-major (for Q * K^T).
//   USE_TDM : stage tiles with the Tensor Data Mover, double-buffered LDS.
// 256 threads = 8 wave32; block tile 128x128x32; each wave owns a 64x32 tile
// = 4x2 v_wmma_f32_16x16x32_bf16 accumulators.
// ---------------------------------------------------------------------------
constexpr int BM = 128, BN = 128, BK = 32;
constexpr int LDA_S = BK + 8;   // 40 bf16 = 80 B rows: conflict-free A frags
constexpr int LDB_S = BN + 8;   // 136 bf16 = 272 B rows

template <typename TOut, bool BT, bool HAS_BIAS, bool USE_TDM>
__global__ __launch_bounds__(256) void gemm_wmma(
    const bf16* __restrict__ A, long long strideA, int lda,
    const bf16* __restrict__ Bm, long long strideB, int ldb,
    TOut* __restrict__ C, long long strideC, int ldc,
    const float* __restrict__ bias, int K)
{
    constexpr bool kTDM = USE_TDM && (HAVE_TDM != 0) && !BT;
    constexpr int NBUF = kTDM ? 2 : 1;
    __shared__ bf16 As[NBUF * BM * LDA_S];
    __shared__ bf16 Bs[NBUF * BK * LDB_S];

    const int tid  = threadIdx.x;
    const int lane = tid & 31;
    const int wave = tid >> 5;      // 0..7
    const int wrow = wave >> 2;     // 0..1 -> 64 rows
    const int wcol = wave & 3;      // 0..3 -> 32 cols
    const int half = lane >> 4;
    const int l16  = lane & 15;

    const int m0 = blockIdx.y * BM;
    const int n0 = blockIdx.x * BN;
    A  += (long long)blockIdx.z * strideA;
    Bm += (long long)blockIdx.z * strideB;
    C  += (long long)blockIdx.z * strideC;

    v8f acc[4][2];
#pragma unroll
    for (int i = 0; i < 4; ++i)
#pragma unroll
        for (int j = 0; j < 2; ++j)
#pragma unroll
            for (int e = 0; e < 8; ++e) acc[i][j][e] = 0.0f;

    // ---- manual staging: 8-byte chunks, register-prefetch pipeline ----
    unsigned long long ra[4], rb[4];
    auto load_regs = [&](int k0) {
#pragma unroll
        for (int u = 0; u < 4; ++u) {
            const int c = tid + u * 256;
            { const int r = c >> 3, col = (c & 7) * 4;   // A: 8 chunks / row
              ra[u] = *(const unsigned long long*)
                          (A + (long long)(m0 + r) * lda + (k0 + col)); }
            if constexpr (!BT) {
                const int r = c >> 5, col = (c & 31) * 4; // B: 32 chunks / row
                rb[u] = *(const unsigned long long*)
                            (Bm + (long long)(k0 + r) * ldb + (n0 + col));
            } else {                                      // B^T: along K
                const int n = c >> 3, kk = (c & 7) * 4;
                rb[u] = *(const unsigned long long*)
                            (Bm + (long long)(n0 + n) * ldb + (k0 + kk));
            }
        }
    };
    auto store_lds = [&]() {
#pragma unroll
        for (int u = 0; u < 4; ++u) {
            const int c = tid + u * 256;
            { const int r = c >> 3, col = (c & 7) * 4;
              *(unsigned long long*)&As[r * LDA_S + col] = ra[u]; }
            if constexpr (!BT) {
                const int r = c >> 5, col = (c & 31) * 4;
                *(unsigned long long*)&Bs[r * LDB_S + col] = rb[u];
            } else {                                      // transpose into LDS
                const int n = c >> 3, kk = (c & 7) * 4;
                bf16 t[4];
                *(unsigned long long*)t = rb[u];
#pragma unroll
                for (int e = 0; e < 4; ++e) Bs[(kk + e) * LDB_S + n] = t[e];
            }
        }
    };

    auto issue_tile = [&](int bufi, int k0) {
        // A tile: 128 rows x 32 elems; pad 4 DW every 16 DW  (80 B rows)
        tdm_load_tile(A + (long long)m0 * lda + k0, lda,
                      &As[bufi * BM * LDA_S], BK, BM, 3u, 3u);
        // B tile: 32 rows x 128 elems; pad 4 DW every 64 DW (272 B rows)
        tdm_load_tile(Bm + (long long)k0 * ldb + n0, ldb,
                      &Bs[bufi * BK * LDB_S], BN, BK, 5u, 3u);
    };

    auto compute = [&](int bufi) {
        const bf16* Ab = &As[bufi * BM * LDA_S];
        const bf16* Bb = &Bs[bufi * BK * LDB_S];
        v16bf af[4], bfb[2];
#pragma unroll
        for (int i = 0; i < 4; ++i) {
            const bf16* ap = &Ab[(wrow * 64 + i * 16 + l16) * LDA_S];
#pragma unroll
            for (int j = 0; j < 8; ++j) {     // VGPR j holds K pair
                const int kk = ((j < 4) ? 2 * j : 16 + 2 * (j - 4)) + half * 8;
                af[i][2 * j]     = ap[kk];
                af[i][2 * j + 1] = ap[kk + 1];
            }
        }
#pragma unroll
        for (int j = 0; j < 2; ++j) {         // lane = col l16, elems = K (table order)
            const bf16* bp = &Bb[wcol * 32 + j * 16 + l16];
#pragma unroll
            for (int e = 0; e < 16; ++e) bfb[j][e] = bp[((e < 8) ? half * 8 + e : 16 + half * 8 + (e - 8)) * LDB_S];
        }
#pragma unroll
        for (int i = 0; i < 4; ++i)
#pragma unroll
            for (int j = 0; j < 2; ++j)
                acc[i][j] = __builtin_amdgcn_wmma_f32_16x16x32_bf16(
                    false, af[i], false, bfb[j], (short)0, acc[i][j],
                    false, false);
    };

    if constexpr (kTDM) {
        int buf = 0;
        if (wave == 0) issue_tile(0, 0);
        for (int k0 = 0; k0 < K; k0 += BK) {
            if (wave == 0) {
                if (k0 + BK < K) { issue_tile(buf ^ 1, k0 + BK); wait_tensorcnt_2(); }
                else             { wait_tensorcnt_0(); }
            }
            __syncthreads();          // tile `buf` visible to all waves
            compute(buf);
            __syncthreads();          // all reads done before buf is re-filled
            buf ^= 1;
        }
    } else {
        load_regs(0);
        for (int k0 = 0; k0 < K; k0 += BK) {
            __syncthreads();          // prior compute done reading LDS
            store_lds();
            __syncthreads();          // LDS tile ready
            if (k0 + BK < K) {
                load_regs(k0 + BK);   // overlaps with the WMMAs below
                if (k0 + 2 * BK < K) {
                    __builtin_prefetch(A + (long long)m0 * lda + k0 + 2 * BK, 0, 1);
                    __builtin_prefetch(Bm + (long long)(BT ? n0 : k0 + 2 * BK) * ldb
                                          + (BT ? k0 + 2 * BK : n0), 0, 1);
                }
            }
            compute(0);
        }
    }

    // ---- epilogue: C/D layout: VGPR v, lane -> row +v+8*half, col l16 ----
#pragma unroll
    for (int i = 0; i < 4; ++i) {
#pragma unroll
        for (int j = 0; j < 2; ++j) {
            const int nn = n0 + wcol * 32 + j * 16 + l16;
#pragma unroll
            for (int v = 0; v < 8; ++v) {
                const int mm = m0 + wrow * 64 + i * 16 + v + half * 8;
                float r = acc[i][j][v];
                if constexpr (HAS_BIAS) r += bias[mm];
                C[(long long)mm * ldc + nn] = (TOut)r;
            }
        }
    }
}

// ---------------------------------------------------------------------------
// Vectorized fp32 -> bf16 conversion: 8 elements / thread (16 B stores).
// ---------------------------------------------------------------------------
__global__ __launch_bounds__(256) void f32_to_bf16(
    const float* __restrict__ in, bf16* __restrict__ out, long long n)
{
    const long long i = ((long long)blockIdx.x * 256 + threadIdx.x) * 8;
    if (i + 8 <= n) {
        const float4 f0 = *(const float4*)(in + i);
        const float4 f1 = *(const float4*)(in + i + 4);
        v8bf v;
        v[0] = (bf16)f0.x; v[1] = (bf16)f0.y; v[2] = (bf16)f0.z; v[3] = (bf16)f0.w;
        v[4] = (bf16)f1.x; v[5] = (bf16)f1.y; v[6] = (bf16)f1.z; v[7] = (bf16)f1.w;
        *(v8bf*)(out + i) = v;
    }
}

// ---------------------------------------------------------------------------
// Row softmax over 512 columns, scaled; fp32 in, bf16 out. One block per row.
// ---------------------------------------------------------------------------
__global__ __launch_bounds__(256) void softmax_rows(
    const float* __restrict__ S, bf16* __restrict__ P, float scale)
{
    __shared__ float red[256];
    const int t = threadIdx.x;
    const float* s = S + (long long)blockIdx.x * 512;
    bf16*       p = P + (long long)blockIdx.x * 512;

    const float a0 = s[t] * scale;
    const float a1 = s[t + 256] * scale;

    red[t] = fmaxf(a0, a1);
    __syncthreads();
    for (int off = 128; off > 0; off >>= 1) {
        if (t < off) red[t] = fmaxf(red[t], red[t + off]);
        __syncthreads();
    }
    const float m = red[0];
    __syncthreads();

    const float e0 = __expf(a0 - m), e1 = __expf(a1 - m);
    red[t] = e0 + e1;
    __syncthreads();
    for (int off = 128; off > 0; off >>= 1) {
        if (t < off) red[t] += red[t + off];
        __syncthreads();
    }
    const float inv = 1.0f / red[0];
    p[t]       = (bf16)(e0 * inv);
    p[t + 256] = (bf16)(e1 * inv);
}

// ---------------------------------------------------------------------------
extern "C" void kernel_launch(void* const* d_in, const int* in_sizes, int n_in,
                              void* d_out, int out_size, void* d_ws, size_t ws_size,
                              hipStream_t stream)
{
    const float* x      = (const float*)d_in[0];   // (B, C, H, W)
    const float* w_qkv  = (const float*)d_in[1];   // (3C, C)
    const float* b_qkv  = (const float*)d_in[2];   // (3C,)
    const float* w_proj = (const float*)d_in[3];   // (C, C)
    const float* b_proj = (const float*)d_in[4];   // (C,)
    float*       out    = (float*)d_out;           // (B, C, H, W) fp32

    constexpr long long Bb = 8, C = 512, N = 96 * 96;   // N = 9216

    // workspace layout (all sizes multiples of 256 B)
    char*  ws   = (char*)d_ws;
    bf16*  qkv  = (bf16*)ws;                         // B x 3C x N
    size_t off  = (size_t)Bb * 3 * C * N * sizeof(bf16);
    float* Smat = (float*)(ws + off); off += (size_t)Bb * C * C * sizeof(float);
    bf16*  Pmat = (bf16*) (ws + off); off += (size_t)Bb * C * C * sizeof(bf16);
    bf16*  O    = (bf16*) (ws + off); off += (size_t)Bb * C * N * sizeof(bf16);
    bf16*  xb   = (bf16*) (ws + off); off += (size_t)Bb * C * N * sizeof(bf16);
    bf16*  wqb  = (bf16*) (ws + off); off += (size_t)3 * C * C * sizeof(bf16);
    bf16*  wpb  = (bf16*) (ws + off);

    const dim3 blk(256);

    // 0) convert fp32 inputs to bf16 once
    f32_to_bf16<<<dim3((unsigned)((Bb * C * N) / 2048)), blk, 0, stream>>>(
        x, xb, Bb * C * N);
    f32_to_bf16<<<dim3((unsigned)((3 * C * C) / 2048)), blk, 0, stream>>>(
        w_qkv, wqb, 3 * C * C);
    f32_to_bf16<<<dim3((unsigned)((C * C) / 2048)), blk, 0, stream>>>(
        w_proj, wpb, C * C);

    // 1) qkv = w_qkv * x + b_qkv         (1536x9216, K=512) -- TDM staged
    gemm_wmma<bf16, false, true, true>
        <<<dim3(N / BN, 3 * C / BM, Bb), blk, 0, stream>>>(
            wqb, 0LL, (int)C,
            xb, C * N, (int)N,
            qkv, 3 * C * N, (int)N,
            b_qkv, (int)C);

    // 2) S = Q * K^T                     (512x512, K=9216; B transposed)
    gemm_wmma<float, true, false, false>
        <<<dim3(C / BN, C / BM, Bb), blk, 0, stream>>>(
            qkv,         3 * C * N, (int)N,     // Q
            qkv + C * N, 3 * C * N, (int)N,     // K (as N x K matrix)
            Smat, C * C, (int)C,
            nullptr, (int)N);

    // 3) P = softmax(S / sqrt(C))
    softmax_rows<<<dim3((unsigned)(Bb * C)), blk, 0, stream>>>(
        Smat, Pmat, 0.0441941738f /* 1/sqrt(512) */);

    // 4) O = P * V                       (512x9216, K=512) -- TDM staged
    gemm_wmma<bf16, false, false, true>
        <<<dim3(N / BN, C / BM, Bb), blk, 0, stream>>>(
            Pmat, C * C, (int)C,
            qkv + 2 * C * N, 3 * C * N, (int)N, // V
            O, C * N, (int)N,
            nullptr, (int)C);

    // 5) out = w_proj * O + b_proj       (512x9216, K=512, fp32) -- TDM staged
    gemm_wmma<float, false, true, true>
        <<<dim3(N / BN, C / BM, Bb), blk, 0, stream>>>(
            wpb, 0LL, (int)C,
            O, C * N, (int)N,
            out, C * N, (int)N,
            b_proj, (int)C);
}
